// SymmetricInfoNCEWithT2TLoss_46016279610195
// MI455X (gfx1250) — compile-verified, matrix operand order checked by reading, not run on hardware
//
#include <hip/hip_runtime.h>
#include <hip/hip_bf16.h>
#include <math.h>

// ---------------------------------------------------------------------------
// Symmetric InfoNCE + T2T loss for MI455X (gfx1250, wave32, WMMA + TDM).
//
// GEMM1: img(4096x1024) x txt_v0^T        -> row/col exp-sums + diagonal
// GEMM2: txt_v0(4096x1024) x txt_other^T  -> row exp-sums + block-diagonal
// Logits bounded by 1/T = 14.28..., so exp() never overflows f32 and
// log-sum-exp needs no max pass -> single streaming pass with f32 atomics.
// Tiles are staged into LDS by the Tensor Data Mover (double-buffered), with
// D# padding (36-dword LDS row stride) for conflict-free b128 fragment reads.
// ---------------------------------------------------------------------------

typedef __attribute__((ext_vector_type(16))) __bf16       v16bf;
typedef __attribute__((ext_vector_type(8)))  float        v8f;
typedef __attribute__((ext_vector_type(4)))  unsigned int v4u;
typedef __attribute__((ext_vector_type(8)))  int          v8i;
typedef __attribute__((ext_vector_type(4)))  int          v4i;

#define NB      4096
#define DIMD    1024
#define INV_T   (1.0f / 0.07f)

#define BM 128
#define BN 128
#define BK 32
#define LROW 36   // f32 per LDS row: 32 data + 4 pad dwords (144 B, 16B aligned)

#if __has_builtin(__builtin_amdgcn_tensor_load_to_lds)
  #define USE_TDM 1
  #if __has_include(<hip/amd_detail/amd_gfx1250_TDM.h>)
    #define TDM_ARITY6 1   // amdgpu-toolchain / therock headers -> 6-arg builtin
  #endif
#endif

#ifdef USE_TDM
// Issue one TDM load of a (BK x 128-row) f32 tile into LDS with padding.
__device__ __forceinline__ void tdm_load_tile(const float* gbase, unsigned rowBase,
                                              unsigned k0, unsigned tensorRows,
                                              void* ldsDst)
{
    unsigned long long ga = (unsigned long long)(uintptr_t)gbase +
        ((unsigned long long)rowBase * DIMD + k0) * 4ull;
    unsigned ldsOff = (unsigned)(uintptr_t)ldsDst;   // low 32 bits = LDS offset

    v4u g0;
    g0.x = 1u;                                        // count=1, user descriptor
    g0.y = ldsOff;                                    // lds_addr
    g0.z = (unsigned)ga;                              // global_addr[31:0]
    g0.w = (unsigned)((ga >> 32) & 0x01FFFFFFull) |   // global_addr[56:32]
           (2u << 30);                                // type=2 ("image")

    v8i g1;
    // data_size=2 (4B), pad_enable, pad_interval enc4 (32 dw), pad_amount enc3 (4 dw)
    g1[0] = (int)((2u << 16) | (1u << 20) | (4u << 22) | (3u << 25));
    g1[1] = (int)((DIMD & 0xFFFFu) << 16);            // tensor_dim0[15:0] (=1024)
    g1[2] = (int)(((DIMD >> 16) & 0xFFFFu) |          // tensor_dim0[31:16]
                  ((tensorRows & 0xFFFFu) << 16));    // tensor_dim1[15:0]
    g1[3] = (int)(((tensorRows >> 16) & 0xFFFFu) |    // tensor_dim1[31:16]
                  ((unsigned)BK << 16));              // tile_dim0 = 32
    g1[4] = (int)128u;                                // tile_dim1 = 128, tile_dim2=0
    g1[5] = (int)DIMD;                                // tensor_dim0_stride = 1024
    g1[6] = 0;                                        // stride hi / dim1_stride
    g1[7] = 0;

    v4i gz = {0, 0, 0, 0};
#ifdef TDM_ARITY6
    v8i gz8 = {0, 0, 0, 0, 0, 0, 0, 0};
    __builtin_amdgcn_tensor_load_to_lds(g0, g1, gz, gz, gz8, 0);
#else
    __builtin_amdgcn_tensor_load_to_lds(g0, g1, gz, gz, 0);
#endif
}
#endif

// Build a v16bf WMMA fragment from an f32 LDS row, converting on the fly.
// 16-bit A/B layout: lanes<16 hold K {half*8..+7, 16+half*8..+7} pairs.
__device__ __forceinline__ v16bf build_frag(const float* rowp, int half)
{
    const float* s0 = rowp + half * 8;
    const float* s1 = rowp + 16 + half * 8;
    float4 a0 = *reinterpret_cast<const float4*>(s0);
    float4 a1 = *reinterpret_cast<const float4*>(s0 + 4);
    float4 a2 = *reinterpret_cast<const float4*>(s1);
    float4 a3 = *reinterpret_cast<const float4*>(s1 + 4);
    union { __hip_bfloat16 h[16]; v16bf v; } u;
    u.h[0]  = __float2bfloat16(a0.x); u.h[1]  = __float2bfloat16(a0.y);
    u.h[2]  = __float2bfloat16(a0.z); u.h[3]  = __float2bfloat16(a0.w);
    u.h[4]  = __float2bfloat16(a1.x); u.h[5]  = __float2bfloat16(a1.y);
    u.h[6]  = __float2bfloat16(a1.z); u.h[7]  = __float2bfloat16(a1.w);
    u.h[8]  = __float2bfloat16(a2.x); u.h[9]  = __float2bfloat16(a2.y);
    u.h[10] = __float2bfloat16(a2.z); u.h[11] = __float2bfloat16(a2.w);
    u.h[12] = __float2bfloat16(a3.x); u.h[13] = __float2bfloat16(a3.y);
    u.h[14] = __float2bfloat16(a3.z); u.h[15] = __float2bfloat16(a3.w);
    return u.v;
}

__global__ __launch_bounds__(256)
void zero_ws_kernel(float* __restrict__ ws, int n) {
    int i = blockIdx.x * blockDim.x + threadIdx.x;
    if (i < n) ws[i] = 0.0f;
}

// mode 0: GEMM1 (colsum + exact diagonal store)
// mode 1: GEMM2 (block-diagonal logit accumulation into `diag`, no colsum)
__global__ __launch_bounds__(256)
void nce_gemm_kernel(const float* __restrict__ X, const float* __restrict__ Y,
                     int yRows,
                     float* __restrict__ rowsum, float* __restrict__ colsum,
                     float* __restrict__ diag, int mode)
{
    __shared__ alignas(16) float Asl[2][BM][LROW];   // 2 x 18 KB
    __shared__ alignas(16) float Bsl[2][BN][LROW];   // 2 x 18 KB

    const int t    = threadIdx.x;
    const int lane = t & 31;
    const int w    = t >> 5;      // wave 0..7
    const int wm   = w >> 2;      // 0..1  (64 rows per wave)
    const int wn   = w & 3;       // 0..3  (32 cols per wave)
    const int half = lane >> 4;
    const int l16  = lane & 15;

    const int rowBase = blockIdx.y * BM;
    const int colBase = blockIdx.x * BN;

    v8f acc[4][2];
    {
        v8f z = {};
        #pragma unroll
        for (int mt = 0; mt < 4; ++mt)
            #pragma unroll
            for (int nt = 0; nt < 2; ++nt)
                acc[mt][nt] = z;
    }

#ifdef USE_TDM
    // ---- async staging via Tensor Data Mover, double buffered -----------
    if (w == 0) tdm_load_tile(X, (unsigned)rowBase, 0u, (unsigned)NB,    &Asl[0][0][0]);
    if (w == 1) tdm_load_tile(Y, (unsigned)colBase, 0u, (unsigned)yRows, &Bsl[0][0][0]);
#else
    const int srow   = t >> 1;
    const int schunk = (t & 1) * 16;
    {
        const float4* px = reinterpret_cast<const float4*>(
            X + (size_t)(rowBase + srow) * DIMD + schunk);
        const float4* py = reinterpret_cast<const float4*>(
            Y + (size_t)(colBase + srow) * DIMD + schunk);
        #pragma unroll
        for (int v = 0; v < 4; ++v) {
            *reinterpret_cast<float4*>(&Asl[0][srow][schunk + v * 4]) = px[v];
            *reinterpret_cast<float4*>(&Bsl[0][srow][schunk + v * 4]) = py[v];
        }
    }
#endif

    int p = 0;
    for (int k0 = 0; k0 < DIMD; k0 += BK) {
#ifdef USE_TDM
        if (w < 2) __builtin_amdgcn_s_wait_tensorcnt(0);
#endif
        __syncthreads();   // buffer p ready for everyone; buffer p^1 free

        if (k0 + BK < DIMD) {
#ifdef USE_TDM
            if (w == 0) tdm_load_tile(X, (unsigned)rowBase, (unsigned)(k0 + BK),
                                      (unsigned)NB,    &Asl[p ^ 1][0][0]);
            if (w == 1) tdm_load_tile(Y, (unsigned)colBase, (unsigned)(k0 + BK),
                                      (unsigned)yRows, &Bsl[p ^ 1][0][0]);
#else
            const float4* px = reinterpret_cast<const float4*>(
                X + (size_t)(rowBase + srow) * DIMD + (k0 + BK) + schunk);
            const float4* py = reinterpret_cast<const float4*>(
                Y + (size_t)(colBase + srow) * DIMD + (k0 + BK) + schunk);
            #pragma unroll
            for (int v = 0; v < 4; ++v) {
                *reinterpret_cast<float4*>(&Asl[p ^ 1][srow][schunk + v * 4]) = px[v];
                *reinterpret_cast<float4*>(&Bsl[p ^ 1][srow][schunk + v * 4]) = py[v];
            }
#endif
        }

        // ---- compute on buffer p: bf16 conversion co-executes with WMMA ----
        v16bf afrag[4], bfrag[2];
        #pragma unroll
        for (int mt = 0; mt < 4; ++mt)
            afrag[mt] = build_frag(&Asl[p][wm * 64 + mt * 16 + l16][0], half);
        #pragma unroll
        for (int nt = 0; nt < 2; ++nt)
            bfrag[nt] = build_frag(&Bsl[p][wn * 32 + nt * 16 + l16][0], half);

        #pragma unroll
        for (int mt = 0; mt < 4; ++mt)
            #pragma unroll
            for (int nt = 0; nt < 2; ++nt)
                acc[mt][nt] = __builtin_amdgcn_wmma_f32_16x16x32_bf16(
                    false, afrag[mt], false, bfrag[nt],
                    (short)0, acc[mt][nt], false, false);

        p ^= 1;
    }

    // ---- epilogue: streaming softmax partials --------------------------
    float rowAcc[4][8];
    #pragma unroll
    for (int mt = 0; mt < 4; ++mt)
        #pragma unroll
        for (int r = 0; r < 8; ++r)
            rowAcc[mt][r] = 0.0f;
    float colAcc[2] = { 0.0f, 0.0f };

    #pragma unroll
    for (int mt = 0; mt < 4; ++mt) {
        #pragma unroll
        for (int nt = 0; nt < 2; ++nt) {
            const int gcol = colBase + wn * 32 + nt * 16 + l16;
            #pragma unroll
            for (int r = 0; r < 8; ++r) {
                // C layout: VGPR r holds M=r (lanes 0-15) / M=r+8 (lanes 16-31)
                const int grow = rowBase + wm * 64 + mt * 16 + r + 8 * half;
                float logit = acc[mt][nt][r] * INV_T;
                float e = __expf(logit);
                rowAcc[mt][r] += e;
                colAcc[nt]    += e;
                if (mode == 0) {
                    if (grow == gcol) diag[grow] = logit;          // unique writer
                } else {
                    if ((gcol >> 2) == grow) atomicAdd(&diag[grow], logit);
                }
            }
        }
    }

    // row sums: reduce across the 16-lane N group, one atomic per row
    #pragma unroll
    for (int mt = 0; mt < 4; ++mt) {
        #pragma unroll
        for (int r = 0; r < 8; ++r) {
            float rs = rowAcc[mt][r];
            rs += __shfl_xor(rs, 8);
            rs += __shfl_xor(rs, 4);
            rs += __shfl_xor(rs, 2);
            rs += __shfl_xor(rs, 1);
            if (l16 == 0)
                atomicAdd(&rowsum[rowBase + wm * 64 + mt * 16 + r + 8 * half], rs);
        }
    }

    // col sums: lane and lane+16 share a column; fold halves, one atomic
    if (mode == 0) {
        #pragma unroll
        for (int nt = 0; nt < 2; ++nt) {
            float cs = colAcc[nt];
            cs += __shfl_xor(cs, 16);
            if (half == 0)
                atomicAdd(&colsum[colBase + wn * 32 + nt * 16 + l16], cs);
        }
    }
}

__global__ __launch_bounds__(256)
void finalize_kernel(const float* __restrict__ rowsum1, const float* __restrict__ colsum1,
                     const float* __restrict__ diag1,   const float* __restrict__ rowsum2,
                     const float* __restrict__ diagsum2, float* __restrict__ out)
{
    __shared__ float s1s[256];
    __shared__ float s2s[256];
    float s1 = 0.0f, s2 = 0.0f;
    for (int i = threadIdx.x; i < NB; i += 256) {
        s1 += 0.5f * (__logf(rowsum1[i]) + __logf(colsum1[i])) - diag1[i];
        s2 += __logf(rowsum2[i]) - 0.25f * diagsum2[i];
    }
    s1s[threadIdx.x] = s1;
    s2s[threadIdx.x] = s2;
    __syncthreads();
    for (int s = 128; s > 0; s >>= 1) {
        if (threadIdx.x < (unsigned)s) {
            s1s[threadIdx.x] += s1s[threadIdx.x + s];
            s2s[threadIdx.x] += s2s[threadIdx.x + s];
        }
        __syncthreads();
    }
    if (threadIdx.x == 0)
        out[0] = s1s[0] / (float)NB + 0.5f * (s2s[0] / (float)NB);
}

extern "C" void kernel_launch(void* const* d_in, const int* in_sizes, int n_in,
                              void* d_out, int out_size, void* d_ws, size_t ws_size,
                              hipStream_t stream) {
    (void)in_sizes; (void)n_in; (void)out_size; (void)ws_size;
    const float* img = (const float*)d_in[0];              // (4096,1024)
    const float* txt = (const float*)d_in[1];              // (4096,1024)
    const float* oth = (const float*)d_in[2];              // (4096,4,1024) == (16384,1024)

    float* ws       = (float*)d_ws;
    float* rowsum1  = ws;                // 4096
    float* colsum1  = ws + 4096;         // 4096
    float* diag1    = ws + 8192;         // 4096
    float* rowsum2  = ws + 12288;        // 4096
    float* diagsum2 = ws + 16384;        // 4096 (80 KB total)

    zero_ws_kernel<<<(5 * NB + 255) / 256, 256, 0, stream>>>(ws, 5 * NB);

    nce_gemm_kernel<<<dim3(NB / BN, NB / BM), dim3(256), 0, stream>>>(
        img, txt, NB, rowsum1, colsum1, diag1, 0);

    nce_gemm_kernel<<<dim3((NB * 4) / BN, NB / BM), dim3(256), 0, stream>>>(
        txt, oth, NB * 4, rowsum2, nullptr, diagsum2, 1);

    finalize_kernel<<<1, 256, 0, stream>>>(rowsum1, colsum1, diag1, rowsum2,
                                           diagsum2, (float*)d_out);
}
